// HGATLayer_71253507440792
// MI455X (gfx1250) — matrix operand order repaired and checked by reading, not compiled, not run on hardware
//
#include <hip/hip_runtime.h>

typedef __attribute__((ext_vector_type(16))) _Float16 v16h;
typedef __attribute__((ext_vector_type(8)))  float    v8f;

#define IN_F     128
#define OUT_F    128
#define N_HEADS  2
#define HEAD_C   64
#define SLOPE    0.2f
#define LN_EPS   1e-5f

#define AS 136   // LDS row stride (halves) for A tile (64 x 128)
#define BS 136   // LDS row stride (halves) for B tile, stored N-major (128 x 128)

// ---- order-preserving float <-> u32 encoding for atomicMax on floats ----
__device__ __forceinline__ unsigned fenc(float f) {
  unsigned u = __float_as_uint(f);
  return (u & 0x80000000u) ? ~u : (u | 0x80000000u);
}
__device__ __forceinline__ float fdec(unsigned u) {
  unsigned v = (u & 0x80000000u) ? (u & 0x7FFFFFFFu) : ~u;
  return __uint_as_float(v);
}

// =====================================================================
// GEMM (X[N,128] @ W[128,128]) with fused attention-logit reduction.
// Block = 128 threads = 4 wave32; block computes a 64 x 128 output tile.
// Each wave owns 16 rows and runs 4 K-steps x 8 N-tiles of
// v_wmma_f32_16x16x32_f16. Hout == nullptr -> only logits are produced.
// =====================================================================
__launch_bounds__(128)
__global__ void gemm_attn_kernel(const float* __restrict__ X,
                                 const float* __restrict__ W,
                                 const float* __restrict__ att,   // [2,64]
                                 float* __restrict__ Hout,        // [N,128] or nullptr
                                 float* __restrict__ Aout,        // [N,2]
                                 int N)
{
  __shared__ _Float16 Ash[64 * AS];
  __shared__ _Float16 Bsh[128 * BS];   // N-major: Bsh[n*BS + k]
  __shared__ float    a_tile[128];     // [64 rows][2 heads]

  const int tid  = threadIdx.x;
  const int wave = tid >> 5;
  const int lane = tid & 31;
  const int half = lane >> 4;
  const int l16  = lane & 15;
  const int rowBase = blockIdx.x * 64;

  a_tile[tid] = 0.0f;

  // Stage X tile (64x128) as f16, zero-filled past N.
  for (int idx = tid; idx < 64 * IN_F; idx += 128) {
    int r = idx >> 7, c = idx & 127;
    int row = rowBase + r;
    float xv = (row < N) ? X[(size_t)row * IN_F + c] : 0.0f;
    Ash[r * AS + c] = (_Float16)xv;
  }
  // Stage W (128x128) transposed (N-major) as f16 so B-fragment K-pairs are contiguous.
  for (int idx = tid; idx < IN_F * OUT_F; idx += 128) {
    int k = idx >> 7, n = idx & 127;
    Bsh[n * BS + k] = (_Float16)W[idx];
  }
  __syncthreads();

  const int m0 = wave * 16;
  v8f acc[8] = {};

  #pragma unroll
  for (int kt = 0; kt < 4; ++kt) {
    const int k0 = kt * 32;
    // A fragment (16x32 f16): lane holds row M=l16; K per ISA layout.
    v16h a;
    #pragma unroll
    for (int v = 0; v < 8; ++v) {
      int kk = (v < 4) ? (half * 8 + 2 * v) : (16 + half * 8 + 2 * (v - 4));
      a[2 * v]     = Ash[(m0 + l16) * AS + k0 + kk];
      a[2 * v + 1] = Ash[(m0 + l16) * AS + k0 + kk + 1];
    }
    #pragma unroll
    for (int t = 0; t < 8; ++t) {
      const int n = t * 16 + l16;
      // B fragment (32x16 f16): lane holds col N=l16; K per ISA layout (same as A).
      v16h b;
      #pragma unroll
      for (int v = 0; v < 8; ++v) {
        int kk = k0 + ((v < 4) ? (half * 8 + 2 * v) : (16 + half * 8 + 2 * (v - 4)));
        b[2 * v]     = Bsh[n * BS + kk];
        b[2 * v + 1] = Bsh[n * BS + kk + 1];
      }
      acc[t] = __builtin_amdgcn_wmma_f32_16x16x32_f16(
          false, a, false, b, (short)0, acc[t], false, false);
    }
  }

  // Epilogue: store H (optional) + fused attention logit via LDS float atomics.
  #pragma unroll
  for (int t = 0; t < 8; ++t) {
    const int n  = t * 16 + l16;
    const int hh = n >> 6;                       // head
    const float av = att[hh * HEAD_C + (n & 63)];
    #pragma unroll
    for (int r = 0; r < 8; ++r) {
      const int ml  = m0 + r + 8 * half;         // C/D layout: M = r + 8*(lane/16)
      const int row = rowBase + ml;
      const float hv = acc[t][r];
      if (Hout && row < N) Hout[(size_t)row * OUT_F + n] = hv;
      atomicAdd(&a_tile[ml * 2 + hh], hv * av);  // ds_add_f32
    }
  }
  __syncthreads();
  {
    int ml = tid >> 1;
    int row = rowBase + ml;
    if (row < N) Aout[(size_t)row * 2 + (tid & 1)] = a_tile[tid];
  }
}

// =====================================================================
// Edge phase
// =====================================================================
__global__ void edge_logit_max_kernel(const long long* __restrict__ ei, int E,
                                      const float* __restrict__ a_s,
                                      const float* __restrict__ a_d,
                                      float* __restrict__ el,      // [E,2] raw logits
                                      unsigned* __restrict__ menc) // [Ndst,2] encoded max
{
  int e = blockIdx.x * blockDim.x + threadIdx.x;
  if (e >= E) return;
  long long s = ei[e];
  long long d = ei[E + e];
  #pragma unroll
  for (int h = 0; h < N_HEADS; ++h) {
    float v = a_s[s * 2 + h] + a_d[d * 2 + h];
    v = (v > 0.0f) ? v : SLOPE * v;            // leaky_relu
    el[(size_t)e * 2 + h] = v;
    atomicMax(&menc[d * 2 + h], fenc(v));
  }
}

__global__ void edge_expsum_kernel(const long long* __restrict__ ei, int E,
                                   float* __restrict__ el,          // in: logit, out: exp
                                   const unsigned* __restrict__ menc,
                                   float* __restrict__ z)           // [Ndst,2]
{
  int e = blockIdx.x * blockDim.x + threadIdx.x;
  if (e >= E) return;
  long long d = ei[E + e];
  #pragma unroll
  for (int h = 0; h < N_HEADS; ++h) {
    float mv = fdec(menc[d * 2 + h]);
    float pe = __expf(el[(size_t)e * 2 + h] - mv);
    el[(size_t)e * 2 + h] = pe;
    atomicAdd(&z[d * 2 + h], pe);
  }
}

// One wave per edge: float4 gather of hs[src], 4 f32 atomic adds per lane.
__global__ void scatter_kernel(const long long* __restrict__ ei, int E,
                               const float* __restrict__ hs,   // [Nsrc,128]
                               const float* __restrict__ p,    // [E,2] exp weights
                               const float* __restrict__ z,    // [Ndst,2]
                               float* __restrict__ agg)        // [Ndst,128]
{
  int gid  = blockIdx.x * blockDim.x + threadIdx.x;
  int e    = gid >> 5;
  int lane = gid & 31;
  if (e >= E) return;
  long long s = ei[e];
  long long d = ei[E + e];
  int hh = lane >> 4;                      // cols lane*4..lane*4+3 share one head
  float al = p[(size_t)e * 2 + hh] / (z[d * 2 + hh] + 1e-16f);
  const float4 hv = ((const float4*)hs)[(size_t)s * 32 + lane];
  float* o = agg + (size_t)d * OUT_F + lane * 4;
  atomicAdd(o + 0, hv.x * al);
  atomicAdd(o + 1, hv.y * al);
  atomicAdd(o + 2, hv.z * al);
  atomicAdd(o + 3, hv.w * al);
}

// =====================================================================
// Bias + LayerNorm + ELU, in place, one wave per node (wave32 shuffles).
// =====================================================================
__global__ void ln_elu_kernel(float* __restrict__ out,
                              const float* __restrict__ bias,
                              const float* __restrict__ g,
                              const float* __restrict__ bb,
                              int N)
{
  int gid  = blockIdx.x * blockDim.x + threadIdx.x;
  int node = gid >> 5;
  int lane = gid & 31;
  if (node >= N) return;
  float4 v = ((float4*)out)[(size_t)node * 32 + lane];
  const float4 bi = ((const float4*)bias)[lane];
  v.x += bi.x; v.y += bi.y; v.z += bi.z; v.w += bi.w;

  float s = v.x + v.y + v.z + v.w;
  #pragma unroll
  for (int off = 16; off > 0; off >>= 1) s += __shfl_xor(s, off, 32);
  float mu = s * (1.0f / OUT_F);

  float dx = v.x - mu, dy = v.y - mu, dz = v.z - mu, dw = v.w - mu;
  float q = dx * dx + dy * dy + dz * dz + dw * dw;
  #pragma unroll
  for (int off = 16; off > 0; off >>= 1) q += __shfl_xor(q, off, 32);
  float rs = rsqrtf(q * (1.0f / OUT_F) + LN_EPS);

  const float4 gv = ((const float4*)g)[lane];
  const float4 bv = ((const float4*)bb)[lane];
  float4 y;
  y.x = dx * rs * gv.x + bv.x;
  y.y = dy * rs * gv.y + bv.y;
  y.z = dz * rs * gv.z + bv.z;
  y.w = dw * rs * gv.w + bv.w;
  y.x = (y.x > 0.0f) ? y.x : (__expf(y.x) - 1.0f);
  y.y = (y.y > 0.0f) ? y.y : (__expf(y.y) - 1.0f);
  y.z = (y.z > 0.0f) ? y.z : (__expf(y.z) - 1.0f);
  y.w = (y.w > 0.0f) ? y.w : (__expf(y.w) - 1.0f);
  ((float4*)out)[(size_t)node * 32 + lane] = y;
}

// =====================================================================
extern "C" void kernel_launch(void* const* d_in, const int* in_sizes, int n_in,
                              void* d_out, int out_size, void* d_ws, size_t ws_size,
                              hipStream_t stream)
{
  const float*     x_user      = (const float*)d_in[0];
  const float*     x_item      = (const float*)d_in[1];
  const long long* ei_u2i      = (const long long*)d_in[2];
  const long long* ei_i2u      = (const long long*)d_in[3];
  const float*     w_src_u2i   = (const float*)d_in[4];
  const float*     w_dst_u2i   = (const float*)d_in[5];
  const float*     att_src_u2i = (const float*)d_in[6];
  const float*     att_dst_u2i = (const float*)d_in[7];
  const float*     b_u2i       = (const float*)d_in[8];
  const float*     w_src_i2u   = (const float*)d_in[9];
  const float*     w_dst_i2u   = (const float*)d_in[10];
  const float*     att_src_i2u = (const float*)d_in[11];
  const float*     att_dst_i2u = (const float*)d_in[12];
  const float*     b_i2u       = (const float*)d_in[13];
  const float*     ln_g_user   = (const float*)d_in[14];
  const float*     ln_b_user   = (const float*)d_in[15];
  const float*     ln_g_item   = (const float*)d_in[16];
  const float*     ln_b_item   = (const float*)d_in[17];

  const int NU = in_sizes[0] / IN_F;
  const int NI = in_sizes[1] / IN_F;
  const int E  = in_sizes[2] / 2;

  // ---- workspace carve-up (floats) ----
  float* ws = (float*)d_ws;
  float* hs_u2i  = ws;              ws += (size_t)NU * OUT_F;   // transformed x_user (src of u2i)
  float* hs_i2u  = ws;              ws += (size_t)NI * OUT_F;   // transformed x_item (src of i2u)
  float* a_s_u2i = ws;              ws += (size_t)NU * 2;
  float* a_d_u2i = ws;              ws += (size_t)NI * 2;
  float* a_s_i2u = ws;              ws += (size_t)NI * 2;
  float* a_d_i2u = ws;              ws += (size_t)NU * 2;
  unsigned* m_item = (unsigned*)ws; ws += (size_t)NI * 2;       // m/z block is contiguous
  float*    z_item = ws;            ws += (size_t)NI * 2;
  unsigned* m_user = (unsigned*)ws; ws += (size_t)NU * 2;
  float*    z_user = ws;            ws += (size_t)NU * 2;
  float* p_u2i   = ws;              ws += (size_t)E * 2;        // logits, then exp weights
  float* p_i2u   = ws;              ws += (size_t)E * 2;

  float* out_user = (float*)d_out;                 // tuple order: (out_user, out_item)
  float* out_item = out_user + (size_t)NU * OUT_F;

  // zero aggregation targets + m/z counters (graph-capturable memset nodes)
  hipMemsetAsync(d_out, 0, (size_t)out_size * sizeof(float), stream);
  hipMemsetAsync((void*)m_item, 0,
                 ((size_t)NI * 4 + (size_t)NU * 4) * sizeof(float), stream);

  dim3 blk(128);
  gemm_attn_kernel<<<(NU + 63) / 64, blk, 0, stream>>>(x_user, w_src_u2i, att_src_u2i, hs_u2i,  a_s_u2i, NU);
  gemm_attn_kernel<<<(NI + 63) / 64, blk, 0, stream>>>(x_item, w_dst_u2i, att_dst_u2i, nullptr, a_d_u2i, NI);
  gemm_attn_kernel<<<(NI + 63) / 64, blk, 0, stream>>>(x_item, w_src_i2u, att_src_i2u, hs_i2u,  a_s_i2u, NI);
  gemm_attn_kernel<<<(NU + 63) / 64, blk, 0, stream>>>(x_user, w_dst_i2u, att_dst_i2u, nullptr, a_d_i2u, NU);

  const int tb = 256;
  const int egrid = (E + tb - 1) / tb;
  edge_logit_max_kernel<<<egrid, tb, 0, stream>>>(ei_u2i, E, a_s_u2i, a_d_u2i, p_u2i, m_item);
  edge_logit_max_kernel<<<egrid, tb, 0, stream>>>(ei_i2u, E, a_s_i2u, a_d_i2u, p_i2u, m_user);
  edge_expsum_kernel<<<egrid, tb, 0, stream>>>(ei_u2i, E, p_u2i, m_item, z_item);
  edge_expsum_kernel<<<egrid, tb, 0, stream>>>(ei_i2u, E, p_i2u, m_user, z_user);

  const long long tsc = (long long)E * 32;
  const int sgrid = (int)((tsc + tb - 1) / tb);
  scatter_kernel<<<sgrid, tb, 0, stream>>>(ei_u2i, E, hs_u2i, p_u2i, z_item, out_item);
  scatter_kernel<<<sgrid, tb, 0, stream>>>(ei_i2u, E, hs_i2u, p_i2u, z_user, out_user);

  const int ugrid = (int)(((long long)NU * 32 + tb - 1) / tb);
  const int igrid = (int)(((long long)NI * 32 + tb - 1) / tb);
  ln_elu_kernel<<<ugrid, tb, 0, stream>>>(out_user, b_i2u, ln_g_user, ln_b_user, NU);
  ln_elu_kernel<<<igrid, tb, 0, stream>>>(out_item, b_u2i, ln_g_item, ln_b_item, NI);
}
